// ScaledDotProductAttention_14491219657479
// MI455X (gfx1250) — matrix operand order repaired and checked by reading, not compiled, not run on hardware
//
#include <hip/hip_runtime.h>

typedef __attribute__((ext_vector_type(16))) _Float16 v16h;
typedef __attribute__((ext_vector_type(8)))  _Float16 v8h;
typedef __attribute__((ext_vector_type(8)))  float    v8f;
typedef __attribute__((ext_vector_type(4)))  float    f32x4;

#define SEQ   2048
#define DKC   128
#define DVC   128
#define BM    128      // query rows per workgroup
#define KSTEP 32       // keys per pipeline step
#define NW    8        // waves per workgroup (256 threads, wave32)

// LDS strides in halves (padded for bank spread; byte stride multiple of 16)
#define SKst  136      // K tile: [key 0..31][d 0..127]
#define SVst  40       // V tile transposed: [dv 0..127][key 0..31]
#define PAst  40       // P staging per wave: [m 0..15][k 0..31]

#define WMMA_F16(A, B, C) \
    __builtin_amdgcn_wmma_f32_16x16x32_f16(false, (A), false, (B), (short)0, (C), false, false)

__global__ __launch_bounds__(256)
void attn_fused_kernel(const float* __restrict__ Q, const float* __restrict__ K,
                       const float* __restrict__ V, float* __restrict__ Out,
                       float* __restrict__ Wts)
{
    __shared__ _Float16 ldsK[32 * SKst];
    __shared__ _Float16 ldsV[DVC * SVst];
    __shared__ _Float16 ldsP[NW * 16 * PAst];

    const int tid  = threadIdx.x;
    const int wave = tid >> 5;
    const int lane = tid & 31;
    const int n16  = lane & 15;
    const int hi   = lane >> 4;          // 0: rows j / cols 0-15 half, 1: rows j+8
    const int bh   = blockIdx.y;         // 0..31
    const int r0   = blockIdx.x * BM;
    const int rowBase = r0 + wave * 16;  // wave's first query row
    // wave-uniform causal limit as an SGPR so guards become scalar branches
    // (WMMA must execute with EXEC all-1s)
    const int rowMaxU = __builtin_amdgcn_readfirstlane(rowBase + 15);

    const float scale = 0.08838834764831845f;  // 1/sqrt(128)

    const float* Qb = Q + (size_t)bh * SEQ * DKC;
    const float* Kb = K + (size_t)bh * SEQ * DKC;
    const float* Vb = V + (size_t)bh * SEQ * DVC;
    float* Ob = Out + (size_t)bh * SEQ * DVC;
    float* Wb = Wts + (size_t)bh * SEQ * SEQ;

    // ---------- load Q fragments once, directly in WMMA A-operand layout ----------
    v16h qa[4];
    {
        const float* qrow = Qb + (size_t)(rowBase + n16) * DKC;
        const int b8 = hi * 8;
        #pragma unroll
        for (int c = 0; c < 4; ++c) {
            f32x4 x0 = *(const f32x4*)(qrow + 32*c + b8);
            f32x4 x1 = *(const f32x4*)(qrow + 32*c + b8 + 4);
            f32x4 x2 = *(const f32x4*)(qrow + 32*c + 16 + b8);
            f32x4 x3 = *(const f32x4*)(qrow + 32*c + 16 + b8 + 4);
            #pragma unroll
            for (int e = 0; e < 4; ++e) {
                qa[c][e]      = (_Float16)x0[e];
                qa[c][4 + e]  = (_Float16)x1[e];
                qa[c][8 + e]  = (_Float16)x2[e];
                qa[c][12 + e] = (_Float16)x3[e];
            }
        }
    }

    // score tile for key half h (cols kb+16h .. +15): hoist all B-fragments,
    // then run two independent WMMA chains to shorten the serial dep chain.
    auto scoreTile = [&](int h) -> v8f {
        v16h b[4];
        #pragma unroll
        for (int cc = 0; cc < 4; ++cc) {
            const v8h* bp = (const v8h*)&ldsK[(16*h + n16) * SKst + 32*cc + hi*8];
            v8h lo = bp[0], hv = bp[2];
            b[cc] = __builtin_shufflevector(lo, hv,
                0,1,2,3,4,5,6,7,8,9,10,11,12,13,14,15);
        }
        v8f a0 = {}, a1 = {};
        a0 = WMMA_F16(qa[0], b[0], a0);
        a1 = WMMA_F16(qa[1], b[1], a1);
        a0 = WMMA_F16(qa[2], b[2], a0);
        a1 = WMMA_F16(qa[3], b[3], a1);
        return a0 + a1;
    };

    const int nsteps = (r0 + BM) / KSTEP;   // causal range for this block

    // ---------- Pass 1: per-row max and softmax denominator ----------
    float m8[8], s8[8];
    #pragma unroll
    for (int j = 0; j < 8; ++j) { m8[j] = -3.0e38f; s8[j] = 0.0f; }

    for (int step = 0; step < nsteps; ++step) {
        const int kb = step * KSTEP;
        // cooperative: K tile (32 keys x 128 d) f32 -> f16 into LDS [key][d]
        {
            const int key = tid >> 3;
            const int dp  = (tid & 7) * 16;
            const float* src = Kb + (size_t)(kb + key) * DKC + dp;
            f32x4 a0 = *(const f32x4*)(src + 0);
            f32x4 a1 = *(const f32x4*)(src + 4);
            f32x4 a2 = *(const f32x4*)(src + 8);
            f32x4 a3 = *(const f32x4*)(src + 12);
            v8h h0, h1;
            #pragma unroll
            for (int e = 0; e < 4; ++e) {
                h0[e]   = (_Float16)a0[e];  h0[4+e] = (_Float16)a1[e];
                h1[e]   = (_Float16)a2[e];  h1[4+e] = (_Float16)a3[e];
            }
            v8h* dst = (v8h*)&ldsK[key * SKst + dp];
            dst[0] = h0; dst[1] = h1;
            if (step + 1 < nsteps)
                __builtin_prefetch(src + (size_t)KSTEP * DKC, 0, 1);
        }
        __syncthreads();

        if (kb <= rowMaxU) {                          // scalar branch (SGPR cmp)
            #pragma unroll
            for (int h = 0; h < 2; ++h) {
                if (kb + 16*h > rowMaxU) break;       // scalar branch
                v8f acc = scoreTile(h);
                #pragma unroll
                for (int j = 0; j < 8; ++j) {
                    const int row = rowBase + j + hi*8;
                    const int col = kb + 16*h + n16;
                    if (col <= row) {
                        const float x  = acc[j] * scale;
                        const float mn = fmaxf(m8[j], x);
                        s8[j] = s8[j] * __expf(m8[j] - mn) + __expf(x - mn);
                        m8[j] = mn;
                    }
                }
            }
        }
        __syncthreads();
    }

    // reduce (m,s) across the 16 lanes sharing each row
    #pragma unroll
    for (int j = 0; j < 8; ++j) {
        float m = m8[j], s = s8[j];
        #pragma unroll
        for (int off = 1; off < 16; off <<= 1) {
            const float mo = __shfl_xor(m, off, 16);
            const float so = __shfl_xor(s, off, 16);
            const float mn = fmaxf(m, mo);
            s = s * __expf(m - mn) + so * __expf(mo - mn);
            m = mn;
        }
        m8[j] = m;
        s8[j] = 1.0f / s;   // inverse denominator
    }

    // ---------- Pass 2: normalized weights out + O = P @ V ----------
    v8f oacc[8];
    #pragma unroll
    for (int c = 0; c < 8; ++c) oacc[c] = (v8f){};

    for (int step = 0; step < nsteps; ++step) {
        const int kb = step * KSTEP;
        // cooperative: K tile [key][d] + V tile transposed [dv][key]
        {
            const int key = tid >> 3;
            const int dp  = (tid & 7) * 16;
            const float* src = Kb + (size_t)(kb + key) * DKC + dp;
            f32x4 a0 = *(const f32x4*)(src + 0);
            f32x4 a1 = *(const f32x4*)(src + 4);
            f32x4 a2 = *(const f32x4*)(src + 8);
            f32x4 a3 = *(const f32x4*)(src + 12);
            v8h h0, h1;
            #pragma unroll
            for (int e = 0; e < 4; ++e) {
                h0[e] = (_Float16)a0[e];  h0[4+e] = (_Float16)a1[e];
                h1[e] = (_Float16)a2[e];  h1[4+e] = (_Float16)a3[e];
            }
            v8h* dst = (v8h*)&ldsK[key * SKst + dp];
            dst[0] = h0; dst[1] = h1;

            const float* vs = Vb + (size_t)(kb + key) * DVC + dp;
            f32x4 b0 = *(const f32x4*)(vs + 0);
            f32x4 b1 = *(const f32x4*)(vs + 4);
            f32x4 b2 = *(const f32x4*)(vs + 8);
            f32x4 b3 = *(const f32x4*)(vs + 12);
            #pragma unroll
            for (int e = 0; e < 4; ++e) {
                ldsV[(dp +  0 + e) * SVst + key] = (_Float16)b0[e];
                ldsV[(dp +  4 + e) * SVst + key] = (_Float16)b1[e];
                ldsV[(dp +  8 + e) * SVst + key] = (_Float16)b2[e];
                ldsV[(dp + 12 + e) * SVst + key] = (_Float16)b3[e];
            }
            if (step + 1 < nsteps) {
                __builtin_prefetch(src + (size_t)KSTEP * DKC, 0, 1);
                __builtin_prefetch(vs  + (size_t)KSTEP * DVC, 0, 1);
            }
        }
        __syncthreads();

        const bool live0 = (kb <= rowMaxU);           // wave-uniform (SGPR)
        _Float16* pmine = &ldsP[wave * 16 * PAst];

        #pragma unroll
        for (int h = 0; h < 2; ++h) {
            const bool liveh = (kb + 16*h) <= rowMaxU;
            v8f acc = {};
            if (liveh)
                acc = scoreTile(h);
            const int col = kb + 16*h + n16;
            #pragma unroll
            for (int j = 0; j < 8; ++j) {
                const int row = rowBase + j + hi*8;
                float w = 0.0f;
                if (liveh && col <= row)
                    w = __expf(acc[j] * scale - m8[j]) * s8[j];
                Wb[row * SEQ + col] = w;                            // normalized weight
                if (live0)
                    pmine[(j + hi*8) * PAst + 16*h + n16] = (_Float16)w;
            }
        }

        if (live0) {
            // rearrange staged P (C/D layout in LDS) into A-operand fragment
            const _Float16* pp = &ldsP[wave * 16 * PAst + n16 * PAst + hi*8];
            v8h plo = *(const v8h*)(pp);
            v8h phi = *(const v8h*)(pp + 16);
            v16h pa = __builtin_shufflevector(plo, phi,
                0,1,2,3,4,5,6,7,8,9,10,11,12,13,14,15);
            // hoist all 8 V fragments, then issue 8 independent WMMAs back-to-back
            v16h vf[8];
            #pragma unroll
            for (int c2 = 0; c2 < 8; ++c2) {
                const v8h* vp = (const v8h*)&ldsV[(16*c2 + n16) * SVst + hi*8];
                v8h lo = vp[0], hv = vp[2];
                vf[c2] = __builtin_shufflevector(lo, hv,
                    0,1,2,3,4,5,6,7,8,9,10,11,12,13,14,15);
            }
            #pragma unroll
            for (int c2 = 0; c2 < 8; ++c2)
                oacc[c2] = WMMA_F16(pa, vf[c2], oacc[c2]);
        }
        __syncthreads();
    }

    // ---------- store O (C/D layout: VGPR j -> row j or j+8, col = 16*c2 + n16) ----------
    #pragma unroll
    for (int c2 = 0; c2 < 8; ++c2) {
        #pragma unroll
        for (int j = 0; j < 8; ++j) {
            const int row = rowBase + j + hi*8;
            Ob[(size_t)row * DVC + 16*c2 + n16] = oacc[c2][j];
        }
    }

    // ---------- zero-fill weight columns beyond this block's causal range ----------
    const int cstart = r0 + BM;
    if (cstart < SEQ) {
        const int zc = SEQ - cstart;         // multiple of 128
        const int per_row4 = zc >> 2;
        const int total4 = BM * per_row4;
        const f32x4 z = {0.0f, 0.0f, 0.0f, 0.0f};
        for (int idx = tid; idx < total4; idx += 256) {
            const int row = r0 + idx / per_row4;
            const int col = cstart + (idx % per_row4) * 4;
            *(f32x4*)&Wb[(size_t)row * SEQ + col] = z;
        }
    }
}

extern "C" void kernel_launch(void* const* d_in, const int* in_sizes, int n_in,
                              void* d_out, int out_size, void* d_ws, size_t ws_size,
                              hipStream_t stream) {
    const float* Q = (const float*)d_in[0];
    const float* K = (const float*)d_in[1];
    const float* V = (const float*)d_in[2];
    // d_in[3] is the causal mask; causality is applied analytically in-kernel.
    float* Out = (float*)d_out;                                   // (B,H,S,DV)
    float* Wts = (float*)d_out + (size_t)2 * 16 * 2048 * 128;     // (B,H,S,S)

    dim3 grid(SEQ / BM, 2 * 16);   // 16 row-blocks x 32 (b,h) pairs
    attn_fused_kernel<<<grid, 256, 0, stream>>>(Q, K, V, Out, Wts);
}
